// GenericSelfAttention_3994319586131
// MI455X (gfx1250) — matrix operand order repaired and checked by reading, not compiled, not run on hardware
//
#include <hip/hip_runtime.h>
#include <hip/hip_bf16.h>
#include <cstdint>
#include <cstddef>

// Problem constants (B, T, C, H) = (4, 2048, 1024, 16), D = 64
#define BDIM 4
#define TDIM 2048
#define CDIM 1024
#define HDIM 16
#define DDIM 64
#define MDIM (BDIM * TDIM)   // 8192 rows

typedef __attribute__((ext_vector_type(16))) __bf16 bf16x16;
typedef __attribute__((ext_vector_type(8)))  float  f32x8;
typedef __attribute__((ext_vector_type(4)))  unsigned int u32x4;
typedef __attribute__((ext_vector_type(8)))  int i32x8;
typedef __attribute__((ext_vector_type(4)))  int i32x4;

union BF16Frag {
  bf16x16 v;
  uint4   q[2];
};

__device__ inline f32x8 zero8() {
  f32x8 z;
#pragma unroll
  for (int i = 0; i < 8; ++i) z[i] = 0.0f;
  return z;
}

// Split 16 fp32 values into hi/lo bf16 (Markidis decomposition).
__device__ inline void split16(const float* t, bf16x16& h, bf16x16& l) {
#pragma unroll
  for (int i = 0; i < 16; ++i) {
    __bf16 hi = (__bf16)t[i];
    h[i] = hi;
    l[i] = (__bf16)(t[i] - (float)hi);
  }
}

// A fragment (16x32 bf16): lane = ks*16+m holds row m, K runs [ks*8..+7], [16+ks*8..+7].
__device__ inline bf16x16 loadA_bf16(const __bf16* __restrict__ base, int ld, int lane) {
  const __bf16* p = base + (size_t)(lane & 15) * ld + ((lane >> 4) << 3);
  BF16Frag f;
  f.q[0] = *(const uint4*)(p);
  f.q[1] = *(const uint4*)(p + 16);
  return f.v;
}

// B fragment (32x16): B[k][n] = Src[n0+n][k0+k]; lane = kh*16+n holds K [kh*8..+7], [16+kh*8..+7].
__device__ inline bf16x16 loadB_bf16(const __bf16* __restrict__ base, size_t ld, int lane) {
  const __bf16* p = base + (size_t)(lane & 15) * ld + ((lane >> 4) << 3);
  BF16Frag f;
  f.q[0] = *(const uint4*)(p);
  f.q[1] = *(const uint4*)(p + 16);
  return f.v;
}

// 3-product split-bf16 MMA: c += a*b with ~fp32 accuracy at bf16 WMMA rate.
__device__ inline void mma_split(const bf16x16& ah, const bf16x16& al,
                                 const bf16x16& bh, const bf16x16& bl, f32x8& c) {
  c = __builtin_amdgcn_wmma_f32_16x16x32_bf16(false, ah, false, bh, (short)0, c, false, false);
  c = __builtin_amdgcn_wmma_f32_16x16x32_bf16(false, al, false, bh, (short)0, c, false, false);
  c = __builtin_amdgcn_wmma_f32_16x16x32_bf16(false, ah, false, bl, (short)0, c, false, false);
}

// ---------------------------------------------------------------------------
// TDM: DMA a (tileRows x tileCols) bf16 tile (contiguous cols, row stride
// tensorCols elements) from global into LDS. D# built per CDNA5 ISA §8.3-8.6.
// Tracked by TENSORcnt. EXEC-independent; issue from one wave per workgroup.
// amdgpu-toolchain (clang-23) form: 6 args, extra int32x8 before cpol.
// ---------------------------------------------------------------------------
__device__ inline void tdm_load_bf16_tile(unsigned ldsOff, const __bf16* gsrc,
                                          unsigned tileRows, unsigned tileCols,
                                          unsigned tensorCols, unsigned tensorRows) {
  const unsigned long long ga = (unsigned long long)(uintptr_t)gsrc;
  u32x4 g0;
  g0[0] = 1u;                                    // count=1 (valid), user mode
  g0[1] = ldsOff;                                // lds_addr (bytes)
  g0[2] = (unsigned)ga;                          // global_addr[31:0]
  g0[3] = ((unsigned)(ga >> 32) & 0x01FFFFFFu)   // global_addr[56:32]
          | (2u << 30);                          // type = 2 ("image")
  i32x8 g1;
  g1[0] = 1 << 16;                                                   // data_size=1 -> 2 bytes
  g1[1] = (int)((tensorCols & 0xFFFFu) << 16);                       // tensor_dim0[15:0]
  g1[2] = (int)((tensorCols >> 16) | ((tensorRows & 0xFFFFu) << 16));// dim0 hi | dim1 lo
  g1[3] = (int)((tensorRows >> 16) | (tileCols << 16));              // dim1 hi | tile_dim0
  g1[4] = (int)tileRows;                                             // tile_dim1 (tile_dim2=0)
  g1[5] = (int)tensorCols;                                           // tensor_dim0_stride[31:0]
  g1[6] = 0;
  g1[7] = 0;
  i32x4 gz4;
  gz4[0] = gz4[1] = gz4[2] = gz4[3] = 0;                             // 2D: groups 2/3 unused
  i32x8 gz8;
#pragma unroll
  for (int i = 0; i < 8; ++i) gz8[i] = 0;
  __builtin_amdgcn_tensor_load_to_lds(g0, g1, gz4, gz4, gz8, 0);
}

// ---------------------------------------------------------------------------
// One-time fp32 -> split bf16 hi/lo conversion (x, weights).
// ---------------------------------------------------------------------------
__global__ __launch_bounds__(256)
void split_f32_kernel(const float* __restrict__ src, __bf16* __restrict__ hi,
                      __bf16* __restrict__ lo, int n4) {
  const int i = blockIdx.x * 256 + threadIdx.x;
  if (i >= n4) return;
  const float4 v = ((const float4*)src)[i];
  const float t[4] = {v.x, v.y, v.z, v.w};
#pragma unroll
  for (int j = 0; j < 4; ++j) {
    const __bf16 h = (__bf16)t[j];
    hi[i * 4 + j] = h;
    lo[i * 4 + j] = (__bf16)(t[j] - (float)h);
  }
}

// ---------------------------------------------------------------------------
// GEMM: out = (A @ W^T + bias) * scale. A: 8192x1024, W: 1024x1024 (N x K),
// both pre-split bf16 hi/lo. TDM double-buffers 128x32 A tiles and 64x32 W
// tiles through LDS (wave 0 issues, TENSORcnt + barrier publish), 8 waves
// (4Mx2N, 32x32 each) consume via ds_load fragments -> v_wmma.
// mode 0: fp32 (M x 1024) -> outF     [final projection]
// mode 1: bf16 hi/lo (B,H,T,D)        [Q (scaled 1/sqrt(D)) and K]
// mode 2: bf16 hi/lo (B,H,D,T)        [V transposed for P@V]
// ---------------------------------------------------------------------------
__global__ __launch_bounds__(256)
void gemm_wmma_tdm(const __bf16* __restrict__ Ahi, const __bf16* __restrict__ Alo,
                   const __bf16* __restrict__ Whi, const __bf16* __restrict__ Wlo,
                   const float* __restrict__ bias,
                   float scale, int mode,
                   float* __restrict__ outF,
                   __bf16* __restrict__ outHi, __bf16* __restrict__ outLo) {
  __shared__ __align__(128) __bf16 sA[2][2][128 * 32];  // [buf][hi/lo] 32 KiB
  __shared__ __align__(128) __bf16 sB[2][2][64 * 32];   // [buf][hi/lo] 16 KiB

  const int lane = threadIdx.x & 31;
  const int wave = threadIdx.x >> 5;
  const int wm = wave >> 1, wn = wave & 1;
  const size_t mBlock = (size_t)blockIdx.x * 128;
  const int    nBlock = blockIdx.y * 64;

  const __bf16* aHiT = Ahi + mBlock * CDIM;
  const __bf16* aLoT = Alo + mBlock * CDIM;
  const __bf16* wHiT = Whi + (size_t)nBlock * CDIM;
  const __bf16* wLoT = Wlo + (size_t)nBlock * CDIM;

  f32x8 acc[2][2] = {{zero8(), zero8()}, {zero8(), zero8()}};

  if (wave == 0) {
    tdm_load_bf16_tile((unsigned)(uintptr_t)sA[0][0], aHiT, 128, 32, CDIM, MDIM);
    tdm_load_bf16_tile((unsigned)(uintptr_t)sA[0][1], aLoT, 128, 32, CDIM, MDIM);
    tdm_load_bf16_tile((unsigned)(uintptr_t)sB[0][0], wHiT, 64, 32, CDIM, CDIM);
    tdm_load_bf16_tile((unsigned)(uintptr_t)sB[0][1], wLoT, 64, 32, CDIM, CDIM);
  }

  for (int k0 = 0; k0 < CDIM; k0 += 32) {
    const int cur = (k0 >> 5) & 1;
    if (wave == 0) __builtin_amdgcn_s_wait_tensorcnt(0);
    __syncthreads();  // buffer `cur` filled; buffer cur^1 free to refill
    if (wave == 0 && k0 + 32 < CDIM) {
      const int kn = k0 + 32;
      tdm_load_bf16_tile((unsigned)(uintptr_t)sA[cur ^ 1][0], aHiT + kn, 128, 32, CDIM, MDIM);
      tdm_load_bf16_tile((unsigned)(uintptr_t)sA[cur ^ 1][1], aLoT + kn, 128, 32, CDIM, MDIM);
      tdm_load_bf16_tile((unsigned)(uintptr_t)sB[cur ^ 1][0], wHiT + kn, 64, 32, CDIM, CDIM);
      tdm_load_bf16_tile((unsigned)(uintptr_t)sB[cur ^ 1][1], wLoT + kn, 64, 32, CDIM, CDIM);
    }

    bf16x16 ah[2], al[2], bh[2], bl[2];
#pragma unroll
    for (int i = 0; i < 2; ++i) {
      const int row = wm * 32 + i * 16;
      ah[i] = loadA_bf16(sA[cur][0] + row * 32, 32, lane);
      al[i] = loadA_bf16(sA[cur][1] + row * 32, 32, lane);
    }
#pragma unroll
    for (int j = 0; j < 2; ++j) {
      const int col = wn * 32 + j * 16;
      bh[j] = loadB_bf16(sB[cur][0] + col * 32, 32, lane);
      bl[j] = loadB_bf16(sB[cur][1] + col * 32, 32, lane);
    }
#pragma unroll
    for (int i = 0; i < 2; ++i)
#pragma unroll
      for (int j = 0; j < 2; ++j)
        mma_split(ah[i], al[i], bh[j], bl[j], acc[i][j]);
  }

  const int nl = lane & 15, mh = lane >> 4;
#pragma unroll
  for (int i = 0; i < 2; ++i)
#pragma unroll
    for (int j = 0; j < 2; ++j) {
      const int n = nBlock + wn * 32 + j * 16 + nl;
      const float bv = bias[n];
#pragma unroll
      for (int r = 0; r < 8; ++r) {
        const size_t m = mBlock + (size_t)(wm * 32 + i * 16 + mh * 8 + r);
        const float y = (acc[i][j][r] + bv) * scale;
        if (mode == 0) {
          outF[m * CDIM + n] = y;
        } else {
          const int b = (int)(m >> 11), t = (int)(m & 2047);
          const int h = n >> 6, d = n & 63;
          const size_t idx = (mode == 1)
            ? ((((size_t)b * HDIM + h) * TDIM + t) * DDIM + d)
            : ((((size_t)b * HDIM + h) * DDIM + d) * TDIM + t);
          const __bf16 hi = (__bf16)y;
          outHi[idx] = hi;
          outLo[idx] = (__bf16)(y - (float)hi);
        }
      }
    }
}

// ---------------------------------------------------------------------------
// Flash attention: per (b,h); block = 8 waves x 16 query rows. Each wave
// streams 2048 keys in chunks of 32: S = Q K^T (WMMA), branchless mask,
// online softmax (shfl_xor butterflies within 16-lane halves), P transposed
// through per-wave LDS to A-layout, acc += P V (V stored D x T).
// Output: y split bf16 hi/lo in (B,T,C) for the final projection.
// ---------------------------------------------------------------------------
__global__ __launch_bounds__(256)
void attn_wmma_split(const __bf16* __restrict__ qh, const __bf16* __restrict__ ql,
                     const __bf16* __restrict__ kh, const __bf16* __restrict__ kl,
                     const __bf16* __restrict__ vh, const __bf16* __restrict__ vl,
                     const int* __restrict__ mask,
                     __bf16* __restrict__ yHi, __bf16* __restrict__ yLo) {
  __shared__ __align__(16) float plds[8][16][36];

  const int lane = threadIdx.x & 31;
  const int wave = threadIdx.x >> 5;
  const int bhid = blockIdx.y;
  const int b = bhid >> 4, h = bhid & 15;
  const int q0 = blockIdx.x * 128 + wave * 16;

  const size_t headTD = (size_t)bhid * TDIM * DDIM;
  const __bf16* qhB = qh + headTD + (size_t)q0 * DDIM;
  const __bf16* qlB = ql + headTD + (size_t)q0 * DDIM;
  const __bf16* khB = kh + headTD;
  const __bf16* klB = kl + headTD;
  const __bf16* vhB = vh + headTD;  // (D, T) layout
  const __bf16* vlB = vl + headTD;
  const int* maskB = mask + b * TDIM;

  bf16x16 qAh[2], qAl[2];
#pragma unroll
  for (int c = 0; c < 2; ++c) {
    qAh[c] = loadA_bf16(qhB + c * 32, DDIM, lane);
    qAl[c] = loadA_bf16(qlB + c * 32, DDIM, lane);
  }

  float mstat[8], lstat[8];
  f32x8 acc[4];
#pragma unroll
  for (int r = 0; r < 8; ++r) { mstat[r] = -3.0e38f; lstat[r] = 0.0f; }
#pragma unroll
  for (int d = 0; d < 4; ++d) acc[d] = zero8();

  const int col = lane & 15, mh = lane >> 4;
  float (*P)[36] = plds[wave];

  for (int j0 = 0; j0 < TDIM; j0 += 32) {
    f32x8 s[2] = {zero8(), zero8()};
#pragma unroll
    for (int nt = 0; nt < 2; ++nt) {
      const size_t rowOff = (size_t)(j0 + nt * 16) * DDIM;
#pragma unroll
      for (int c = 0; c < 2; ++c) {
        bf16x16 kbh = loadB_bf16(khB + rowOff + c * 32, DDIM, lane);
        bf16x16 kbl = loadB_bf16(klB + rowOff + c * 32, DDIM, lane);
        mma_split(qAh[c], qAl[c], kbh, kbl, s[nt]);
      }
    }

    const int mv0 = maskB[j0 + col];
    const int mv1 = maskB[j0 + 16 + col];
#pragma unroll
    for (int r = 0; r < 8; ++r) {
      s[0][r] = mv0 ? s[0][r] : -3.0e38f;
      s[1][r] = mv1 ? s[1][r] : -3.0e38f;
    }

    float rmax[8], rsum[8];
#pragma unroll
    for (int r = 0; r < 8; ++r) rmax[r] = fmaxf(s[0][r], s[1][r]);
#pragma unroll
    for (int off = 1; off < 16; off <<= 1)
#pragma unroll
      for (int r = 0; r < 8; ++r) rmax[r] = fmaxf(rmax[r], __shfl_xor(rmax[r], off, 32));

#pragma unroll
    for (int r = 0; r < 8; ++r) {
      const float mnew = fmaxf(mstat[r], rmax[r]);
      const float sc = __expf(mstat[r] - mnew);
      mstat[r] = mnew;
      lstat[r] *= sc;
#pragma unroll
      for (int d = 0; d < 4; ++d) acc[d][r] *= sc;
      s[0][r] = __expf(s[0][r] - mnew);
      s[1][r] = __expf(s[1][r] - mnew);
      rsum[r] = s[0][r] + s[1][r];
    }
#pragma unroll
    for (int off = 1; off < 16; off <<= 1)
#pragma unroll
      for (int r = 0; r < 8; ++r) rsum[r] += __shfl_xor(rsum[r], off, 32);
#pragma unroll
    for (int r = 0; r < 8; ++r) lstat[r] += rsum[r];

    // transpose P (C/D layout -> A layout) through per-wave LDS
#pragma unroll
    for (int r = 0; r < 8; ++r) {
      P[mh * 8 + r][col]      = s[0][r];
      P[mh * 8 + r][16 + col] = s[1][r];
    }
    asm volatile("s_wait_dscnt 0x0" ::: "memory");

    bf16x16 pAh, pAl;
    {
      const float* pp = &P[lane & 15][(lane >> 4) << 3];
      alignas(16) float t[16];
      *(float4*)(t + 0)  = *(const float4*)(pp + 0);
      *(float4*)(t + 4)  = *(const float4*)(pp + 4);
      *(float4*)(t + 8)  = *(const float4*)(pp + 16);
      *(float4*)(t + 12) = *(const float4*)(pp + 20);
      split16(t, pAh, pAl);
    }

#pragma unroll
    for (int dt = 0; dt < 4; ++dt) {
      bf16x16 vbh = loadB_bf16(vhB + (size_t)(dt * 16) * TDIM + j0, TDIM, lane);
      bf16x16 vbl = loadB_bf16(vlB + (size_t)(dt * 16) * TDIM + j0, TDIM, lane);
      mma_split(pAh, pAl, vbh, vbl, acc[dt]);
    }
  }

#pragma unroll
  for (int r = 0; r < 8; ++r) lstat[r] = 1.0f / lstat[r];
#pragma unroll
  for (int dt = 0; dt < 4; ++dt)
#pragma unroll
    for (int r = 0; r < 8; ++r) {
      const int t = q0 + mh * 8 + r;
      const int c = h * DDIM + dt * 16 + col;
      const float yv = acc[dt][r] * lstat[r];
      const size_t idx = ((size_t)b * TDIM + t) * CDIM + c;
      const __bf16 hv = (__bf16)yv;
      yHi[idx] = hv;
      yLo[idx] = (__bf16)(yv - (float)hv);
    }
}

// ---------------------------------------------------------------------------
extern "C" void kernel_launch(void* const* d_in, const int* in_sizes, int n_in,
                              void* d_out, int out_size, void* d_ws, size_t ws_size,
                              hipStream_t stream) {
  const float* x   = (const float*)d_in[0];
  const int*   msk = (const int*)  d_in[1];
  const float* Wq  = (const float*)d_in[2];
  const float* bq  = (const float*)d_in[3];
  const float* Wk  = (const float*)d_in[4];
  const float* bk  = (const float*)d_in[5];
  const float* Wv  = (const float*)d_in[6];
  const float* bv  = (const float*)d_in[7];
  const float* Wp  = (const float*)d_in[8];
  const float* bp  = (const float*)d_in[9];
  float* out = (float*)d_out;

  char* ws = (char*)d_ws;
  const size_t SB = (size_t)MDIM * CDIM * sizeof(__bf16);       // 16 MiB
  const size_t WB = (size_t)CDIM * CDIM * sizeof(__bf16);       // 2 MiB
  __bf16* xHi = (__bf16*)(ws + 0 * SB);
  __bf16* xLo = (__bf16*)(ws + 1 * SB);
  __bf16* qHi = (__bf16*)(ws + 2 * SB);
  __bf16* qLo = (__bf16*)(ws + 3 * SB);
  __bf16* kHi = (__bf16*)(ws + 4 * SB);
  __bf16* kLo = (__bf16*)(ws + 5 * SB);
  __bf16* vHi = (__bf16*)(ws + 6 * SB);
  __bf16* vLo = (__bf16*)(ws + 7 * SB);
  __bf16* yHi = (__bf16*)(ws + 8 * SB);
  __bf16* yLo = (__bf16*)(ws + 9 * SB);
  char* wsW = ws + 10 * SB;
  __bf16* wqHi = (__bf16*)(wsW + 0 * WB);
  __bf16* wqLo = (__bf16*)(wsW + 1 * WB);
  __bf16* wkHi = (__bf16*)(wsW + 2 * WB);
  __bf16* wkLo = (__bf16*)(wsW + 3 * WB);
  __bf16* wvHi = (__bf16*)(wsW + 4 * WB);
  __bf16* wvLo = (__bf16*)(wsW + 5 * WB);
  __bf16* wpHi = (__bf16*)(wsW + 6 * WB);
  __bf16* wpLo = (__bf16*)(wsW + 7 * WB);

  const dim3 blk(256, 1, 1);

  // One-time fp32 -> split bf16 conversions
  const int n4x = MDIM * CDIM / 4;
  const int n4w = CDIM * CDIM / 4;
  split_f32_kernel<<<dim3((n4x + 255) / 256), blk, 0, stream>>>(x,  xHi, xLo, n4x);
  split_f32_kernel<<<dim3((n4w + 255) / 256), blk, 0, stream>>>(Wq, wqHi, wqLo, n4w);
  split_f32_kernel<<<dim3((n4w + 255) / 256), blk, 0, stream>>>(Wk, wkHi, wkLo, n4w);
  split_f32_kernel<<<dim3((n4w + 255) / 256), blk, 0, stream>>>(Wv, wvHi, wvLo, n4w);
  split_f32_kernel<<<dim3((n4w + 255) / 256), blk, 0, stream>>>(Wp, wpHi, wpLo, n4w);

  // Projections (TDM-staged WMMA GEMM): Q pre-scaled 1/sqrt(D); V transposed.
  const dim3 gP(MDIM / 128, CDIM / 64, 1);
  gemm_wmma_tdm<<<gP, blk, 0, stream>>>(xHi, xLo, wqHi, wqLo, bq, 0.125f, 1, nullptr, qHi, qLo);
  gemm_wmma_tdm<<<gP, blk, 0, stream>>>(xHi, xLo, wkHi, wkLo, bk, 1.0f,   1, nullptr, kHi, kLo);
  gemm_wmma_tdm<<<gP, blk, 0, stream>>>(xHi, xLo, wvHi, wvLo, bv, 1.0f,   2, nullptr, vHi, vLo);

  const dim3 gA(TDIM / 128, BDIM * HDIM, 1);
  attn_wmma_split<<<gA, blk, 0, stream>>>(qHi, qLo, kHi, kLo, vHi, vLo, msk, yHi, yLo);

  gemm_wmma_tdm<<<gP, blk, 0, stream>>>(yHi, yLo, wpHi, wpLo, bp, 1.0f, 0, out, nullptr, nullptr);
}
